// TAWLS_Block_62294205661598
// MI455X (gfx1250) — matrix operand order repaired and checked by reading, not compiled, not run on hardware
//
#include <hip/hip_runtime.h>
#include <hip/hip_bf16.h>

// ---------------- types ----------------
typedef __bf16 bf16_t;
typedef __attribute__((ext_vector_type(16))) __bf16 v16bf;
typedef __attribute__((ext_vector_type(8)))  __bf16 v8bf;
typedef __attribute__((ext_vector_type(8)))  float  v8f;
typedef __attribute__((ext_vector_type(4))) unsigned int u32x4;
typedef __attribute__((ext_vector_type(4))) int i32x4;
typedef __attribute__((ext_vector_type(8))) int i32x8;

#define Nn  128
#define Cc  64
#define Tt  64
#define Vv  25
#define HID 32
#define HH  8
#define DD  8
#define QC  192
#define QSCALE 0.35355339059327373f  // 1/sqrt(d=8)

#if __has_builtin(__builtin_amdgcn_tensor_load_to_lds)
#define HAVE_TDM 1
#else
#define HAVE_TDM 0
#endif

static __device__ inline bf16_t to_bf16(float f)   { return (bf16_t)f; }
static __device__ inline float  from_bf16(bf16_t h){ return (float)h;  }

static __device__ inline v8f wmma_bf16(v16bf a, v16bf b, v8f c) {
  return __builtin_amdgcn_wmma_f32_16x16x32_bf16(
      false, a, false, b, (short)0, c, false, false);
}

// load 8 consecutive floats (LDS) as two b128 vector loads
static __device__ inline void ld8f(const float* p, float* out) {
  float4 a = *(const float4*)p;
  float4 b = *(const float4*)(p + 4);
  out[0] = a.x; out[1] = a.y; out[2] = a.z; out[3] = a.w;
  out[4] = b.x; out[5] = b.y; out[6] = b.z; out[7] = b.w;
}

// Fragment-ready layouts (contiguous k order used by this code, wave32):
//   A (16x32, row m, col k): lane = (m&15) + 16*((k&31)>>4); j = k & 15
//   B (32x16, row k, col n): lane = (n&15) + 16*((k&31)>>4); j = k & 15
static __device__ __host__ inline int a_slot(int m, int k, int kTiles) {
  int mt = m >> 4, row = m & 15;
  int ks = k >> 5, kk = k & 31;
  int lane = row + 16 * (kk >> 4);
  int j = kk & 15;
  return ((mt * kTiles + ks) * 32 + lane) * 16 + j;
}
// base slot of the 8-wide K-octet starting at k0 (k0 % 8 == 0), B layout
static __device__ inline int b_oct(int k0, int n, int nTiles) {
  int ks = k0 >> 5, kk = k0 & 31;
  int lane = (n & 15) + 16 * (kk >> 4);
  return ((ks * nTiles + (n >> 4)) * 32 + lane) * 16 + (kk & 15);
}
static __device__ inline v16bf ldfrag(const bf16_t* p) { return *(const v16bf*)p; }

// ============ K0: pre-swizzle weights into fragment-ready layout ============
__global__ void k0_prep(const float* __restrict__ Wt0, const float* __restrict__ Wt1,
                        const float* __restrict__ Wqkv, const float* __restrict__ Wa,
                        bf16_t* __restrict__ fW0, bf16_t* __restrict__ fW1,
                        bf16_t* __restrict__ fWq, bf16_t* __restrict__ fWa) {
  int idx = blockIdx.x * 256 + threadIdx.x;
  if (idx < HID * Cc)  { int m = idx / Cc,  k = idx % Cc;  fW0[a_slot(m, k, 2)] = to_bf16(Wt0[idx]); }
  if (idx < HID * HID) { int m = idx / HID, k = idx % HID; fW1[a_slot(m, k, 1)] = to_bf16(Wt1[idx]); }
  if (idx < QC * Cc)   { int m = idx / Cc,  k = idx % Cc;
                         float w = Wqkv[idx]; if (m < Cc) w *= QSCALE;
                         fWq[a_slot(m, k, 2)] = to_bf16(w); }
  if (idx < Cc * Cc)   { int m = idx / Cc,  k = idx % Cc;  fWa[a_slot(m, k, 2)] = to_bf16(Wa[idx]); }
}

// ================= K1: fused 2-stage MLP (per (n,t)), 128 threads ==========
__global__ void k1_mlp(const float* __restrict__ x,
                       const bf16_t* __restrict__ fW0g, const bf16_t* __restrict__ fW1g,
                       const float* __restrict__ bt0,
                       const float* __restrict__ g0,  const float* __restrict__ be0,
                       const float* __restrict__ m0,  const float* __restrict__ v0,
                       const float* __restrict__ p0,
                       const float* __restrict__ bt1,
                       const float* __restrict__ g1,  const float* __restrict__ be1,
                       const float* __restrict__ m1,  const float* __restrict__ v1,
                       const float* __restrict__ p1,
                       bf16_t* __restrict__ hbuf) {
  __shared__ __align__(32) bf16_t fXS[4 * 512];
  __shared__ __align__(32) bf16_t fH0[2 * 512];
  __shared__ __align__(16) float s0S[HID], d0S[HID], s1S[HID], d1S[HID];

  int tid = threadIdx.x;
  int nt  = blockIdx.x;
  int n = nt / Tt, t = nt % Tt;

  if (tid < HID) {
    float s  = g0[tid] * rsqrtf(v0[tid] + 1e-5f);
    s0S[tid] = s;  d0S[tid] = s * (bt0[tid] - m0[tid]) + be0[tid];
    float s1 = g1[tid] * rsqrtf(v1[tid] + 1e-5f);
    s1S[tid] = s1; d1S[tid] = s1 * (bt1[tid] - m1[tid]) + be1[tid];
  }
  // stage x slice -> B-frag layout, one 8-wide c-octet per task
  for (int i = tid; i < 8 * 32; i += 128) {
    int o = i >> 5, v = i & 31, c0 = o * 8;
    v8bf pk = {};
    if (v < Vv) {
#pragma unroll
      for (int r = 0; r < 8; ++r)
        pk[r] = to_bf16(x[(((size_t)n * Cc + c0 + r) * Tt + t) * Vv + v]);
    }
    *(v8bf*)(fXS + b_oct(c0, v, 2)) = pk;
  }
  __syncthreads();

  float a0 = p0[0], a1 = p1[0];
  int wave = tid >> 5, l = tid & 31;
  int mt = wave >> 1, ntile = wave & 1;
  int hb = 8 * (l >> 4);

  // GEMM1 -> BN+PReLU -> fH0 (masked by 'keep', no branches)
  {
    v8f acc = {};
#pragma unroll
    for (int ks = 0; ks < 2; ++ks) {
      v16bf a = ldfrag(fW0g + ((mt * 2 + ks) * 32 + l) * 16);
      v16bf b = ldfrag(fXS + ((ks * 2 + ntile) * 32 + l) * 16);
      acc = wmma_bf16(a, b, acc);
    }
    int col = ntile * 16 + (l & 15);
    float keep = (col < Vv) ? 1.f : 0.f;
    float sv[8], dv[8];
    ld8f(s0S + mt * 16 + hb, sv);
    ld8f(d0S + mt * 16 + hb, dv);
    v8bf pk;
#pragma unroll
    for (int r = 0; r < 8; ++r) {
      float val = (sv[r] * acc[r] + dv[r]) * keep;
      pk[r] = to_bf16((val >= 0.f) ? val : a0 * val);
    }
    *(v8bf*)(fH0 + (ntile * 32 + (col & 15) + 16 * mt) * 16 + hb) = pk;
  }
  __syncthreads();

  // GEMM2 -> BN+PReLU -> hbuf (oc fast axis: one v8bf global store)
  {
    v16bf a = ldfrag(fW1g + (mt * 32 + l) * 16);
    v16bf b = ldfrag(fH0 + (ntile * 32 + l) * 16);
    v8f acc = {};
    acc = wmma_bf16(a, b, acc);
    int col = ntile * 16 + (l & 15);
    if (col < Vv) {
      float sv[8], dv[8];
      ld8f(s1S + mt * 16 + hb, sv);
      ld8f(d1S + mt * 16 + hb, dv);
      v8bf pk;
#pragma unroll
      for (int r = 0; r < 8; ++r) {
        float val = sv[r] * acc[r] + dv[r];
        pk[r] = to_bf16((val >= 0.f) ? val : a1 * val);
      }
      *(v8bf*)(hbuf + ((size_t)nt * Vv + col) * HID + mt * 16 + hb) = pk;
    }
  }
}

// ========== K2: adjacency + concat + QKV projection (per (n,t)) ==========
__global__ void k2_qkv(const bf16_t* __restrict__ hbuf,
                       const float*  __restrict__ Aw,
                       const bf16_t* __restrict__ fWqg,
                       const float*  __restrict__ bqkv,
                       bf16_t* __restrict__ qkvbuf) {
  __shared__ __align__(32) bf16_t fFS[4 * 512];
  __shared__ __align__(16) bf16_t hS[Vv * HID];   // raw [v][c] (async copied)
  __shared__ __align__(16) float AtS[Vv * Vv];    // raw (TDM copied)
  __shared__ __align__(16) float biasS[QC];

  int tid = threadIdx.x;
  int nt  = blockIdx.x;
  int n = nt / Tt, t = nt % Tt;

  const bf16_t* hp = hbuf + (size_t)nt * (Vv * HID);
  __builtin_prefetch(fWqg, 0, 0);                 // global_prefetch_b8

  // ---- CDNA5 async raw copy: global bf16 -> LDS (1600B = 100 x 16B) ----
  if (tid < 100) {
    unsigned lds_off = (unsigned)(size_t)&hS[tid * 8];
    unsigned long long ga = (unsigned long long)(size_t)(hp + tid * 8);
    asm volatile("global_load_async_to_lds_b128 %0, %1, off"
                 :: "v"(lds_off), "v"(ga) : "memory");
  }

  // ---- CDNA5 Tensor Data Mover: A[t] slice (625 f32, 1D tile) -> LDS ----
#if HAVE_TDM
  {
    unsigned lds_at = (unsigned)(size_t)&AtS[0];
    unsigned long long ga = (unsigned long long)(size_t)(Aw + (size_t)t * Vv * Vv);
    u32x4 g0 = {};
    g0[0] = 1u;                                   // count=1, user mode
    g0[1] = lds_at;                               // lds_addr
    g0[2] = (unsigned)ga;                         // global_addr[31:0]
    g0[3] = (unsigned)(ga >> 32) | (2u << 30);    // global_addr[56:32] | type=2
    i32x8 g1 = {};
    g1[0] = (2 << 16);                            // data_size = 4B
    g1[1] = (int)((625u & 0xffffu) << 16);        // tensor_dim0 lo16
    g1[2] = (int)(1u << 16);                      // tensor_dim1 = 1
    g1[3] = (int)(625u << 16);                    // tile_dim0 = 625
    g1[4] = 0;                                    // tile_dim1/2 unused
    g1[5] = 625;                                  // tensor_dim0_stride
    g1[6] = 0; g1[7] = 0;
    i32x4 g2 = {}, g3 = {};
    if (tid < 32) {                               // one wave issues the DMA
#if __clang_major__ >= 23
      i32x8 g4 = {};
      __builtin_amdgcn_tensor_load_to_lds(g0, g1, g2, g3, g4, 0);
#else
      __builtin_amdgcn_tensor_load_to_lds(g0, g1, g2, g3, 0);
#endif
      __builtin_amdgcn_s_wait_tensorcnt(0);
    }
  }
#else
  for (int i = tid; i < Vv * Vv; i += 256) AtS[i] = Aw[(size_t)t * Vv * Vv + i];
#endif

  for (int i = tid; i < QC;      i += 256) biasS[i] = bqkv[i] * ((i < Cc) ? QSCALE : 1.0f);
  for (int i = tid; i < 4 * 512 / 8; i += 256) { v8bf z = {}; ((v8bf*)fFS)[i] = z; }
  asm volatile("s_wait_asynccnt 0" ::: "memory");
  __syncthreads();

  // f rows 0..31: xu = h * A_t (8-wide c-octet per task, vector LDS I/O)
  for (int i = tid; i < 4 * Vv; i += 256) {
    int o = i / Vv, w = i % Vv, c0 = o * 8;
    float acc[8] = {};
    for (int v = 0; v < Vv; ++v) {
      float aw = AtS[v * Vv + w];
      v8bf hv = *(const v8bf*)(hS + v * HID + c0);
#pragma unroll
      for (int r = 0; r < 8; ++r) acc[r] += from_bf16(hv[r]) * aw;
    }
    v8bf pk;
#pragma unroll
    for (int r = 0; r < 8; ++r) pk[r] = to_bf16(acc[r]);
    *(v8bf*)(fFS + b_oct(c0, w, 2)) = pk;
  }
  // f rows 32..63: copy h (vector octets)
  for (int i = tid; i < 4 * Vv; i += 256) {
    int o = i / Vv, v = i % Vv, c0 = o * 8;
    v8bf hv = *(const v8bf*)(hS + v * HID + c0);
    *(v8bf*)(fFS + b_oct(HID + c0, v, 2)) = hv;
  }
  __syncthreads();

  int wave = tid >> 5, l = tid & 31;
  int hb = 8 * (l >> 4);
  for (int tile = wave; tile < 24; tile += 8) {
    int mt = tile >> 1, ntile = tile & 1;
    v8f acc = {};
#pragma unroll
    for (int ks = 0; ks < 2; ++ks) {
      v16bf a = ldfrag(fWqg + ((mt * 2 + ks) * 32 + l) * 16);
      v16bf b = ldfrag(fFS + ((ks * 2 + ntile) * 32 + l) * 16);
      acc = wmma_bf16(a, b, acc);
    }
    int col = ntile * 16 + (l & 15);
    if (col < Vv) {
      float bv[8];
      ld8f(biasS + mt * 16 + hb, bv);
#pragma unroll
      for (int r = 0; r < 8; ++r) {
        int oc = mt * 16 + hb + r;
        qkvbuf[(((size_t)n * Vv + col) * QC + oc) * Tt + t] = to_bf16(acc[r] + bv[r]);
      }
    }
  }
}

// ============ K3: temporal attention, one wave per (n, v, head) ============
__global__ void k3_attn(const bf16_t* __restrict__ qkvbuf,
                        bf16_t* __restrict__ aobuf) {
  __shared__ __align__(32) bf16_t fragQ[4 * 512];  // A frags (K=8 pad 32)
  __shared__ __align__(32) bf16_t fragK[4 * 512];  // B frags
  __shared__ __align__(32) bf16_t fragV[2 * 512];  // B frags (GEMM2)
  __shared__ __align__(32) bf16_t fragP[8 * 512];  // probs as A frags
  __shared__ __align__(16) float wS[Tt * Tt];

  int l = threadIdx.x;
  int b = blockIdx.x;
  int h = b % HH; int nv = b / HH;
  int v = nv % Vv, n = nv / Vv;
  const bf16_t* base = qkvbuf + ((size_t)(n * Vv + v) * QC) * Tt;

  {  // zero pad slots
    v8bf z = {};
    for (int i = l; i < (4 * 512 + 4 * 512 + 2 * 512) / 8; i += 32) {
      if (i < 256)       ((v8bf*)fragQ)[i] = z;
      else if (i < 512)  ((v8bf*)fragK)[i - 256] = z;
      else               ((v8bf*)fragV)[i - 512] = z;
    }
  }
  __syncthreads();

  // stage q,k: one d-octet (j-contiguous) per row -> single v8bf LDS store
  for (int t = l; t < Tt; t += 32) {
    v8bf pq, pk;
#pragma unroll
    for (int d = 0; d < DD; ++d) {
      pq[d] = base[(h * DD + d) * Tt + t];
      pk[d] = base[(Cc + h * DD + d) * Tt + t];
    }
    *(v8bf*)(fragQ + ((t >> 4) * 32 + (t & 15)) * 16) = pq;
    *(v8bf*)(fragK + ((t >> 4) * 32 + (t & 15)) * 16) = pk;
  }
  // stage v as B^T frags (global rows t-contiguous -> v8bf both sides)
  for (int c = l; c < DD * 8; c += 32) {
    int d = c >> 3, s0 = (c & 7) * 8;
    v8bf val = *(const v8bf*)(base + (2 * Cc + h * DD + d) * Tt + s0);
    *(v8bf*)(fragV + ((s0 >> 5) * 32 + d + 16 * ((s0 >> 4) & 1)) * 16 + (s0 & 15)) = val;
  }
  __syncthreads();

  // logits = q^T k (K=8 zero-padded to 32)
#pragma unroll
  for (int mt = 0; mt < 4; ++mt) {
    v16bf a = ldfrag(fragQ + (mt * 32 + l) * 16);
#pragma unroll
    for (int st = 0; st < 4; ++st) {
      v16bf bf = ldfrag(fragK + (st * 32 + l) * 16);
      v8f acc = {};
      acc = wmma_bf16(a, bf, acc);
#pragma unroll
      for (int r = 0; r < 8; ++r)
        wS[(mt * 16 + r + 8 * (l >> 4)) * Tt + st * 16 + (l & 15)] = acc[r];
    }
  }
  __syncthreads();

  // row softmax (float4 loads); write probs straight into A-frag layout
  for (int rr = l; rr < Tt; rr += 32) {
    const float4* row4 = (const float4*)(wS + rr * Tt);
    float4 R[16];
#pragma unroll
    for (int s4 = 0; s4 < 16; ++s4) R[s4] = row4[s4];
    float mx = -3.4e38f;
#pragma unroll
    for (int s4 = 0; s4 < 16; ++s4)
      mx = fmaxf(mx, fmaxf(fmaxf(R[s4].x, R[s4].y), fmaxf(R[s4].z, R[s4].w)));
    float sum = 0.f;
#pragma unroll
    for (int s4 = 0; s4 < 16; ++s4) {
      R[s4].x = __expf(R[s4].x - mx); R[s4].y = __expf(R[s4].y - mx);
      R[s4].z = __expf(R[s4].z - mx); R[s4].w = __expf(R[s4].w - mx);
      sum += R[s4].x + R[s4].y + R[s4].z + R[s4].w;
    }
    float inv = 1.0f / sum;
#pragma unroll
    for (int o = 0; o < 8; ++o) {
      float4 e0 = R[2 * o], e1 = R[2 * o + 1];
      v8bf pk;
      pk[0] = to_bf16(e0.x * inv); pk[1] = to_bf16(e0.y * inv);
      pk[2] = to_bf16(e0.z * inv); pk[3] = to_bf16(e0.w * inv);
      pk[4] = to_bf16(e1.x * inv); pk[5] = to_bf16(e1.y * inv);
      pk[6] = to_bf16(e1.z * inv); pk[7] = to_bf16(e1.w * inv);
      *(v8bf*)(fragP + a_slot(rr, o * 8, 2)) = pk;
    }
  }
  __syncthreads();

  // ao = probs x V
#pragma unroll
  for (int mt = 0; mt < 4; ++mt) {
    v8f acc = {};
#pragma unroll
    for (int ks = 0; ks < 2; ++ks) {
      v16bf a = ldfrag(fragP + ((mt * 2 + ks) * 32 + l) * 16);
      v16bf bf = ldfrag(fragV + (ks * 32 + l) * 16);
      acc = wmma_bf16(a, bf, acc);
    }
    int d = l & 15;
    if (d < DD) {
      int c = h * DD + d;
#pragma unroll
      for (int r = 0; r < 8; ++r) {
        int t = mt * 16 + r + 8 * (l >> 4);
        aobuf[(((size_t)n * Tt + t) * Cc + c) * Vv + v] = to_bf16(acc[r]);
      }
    }
  }
}

// ====== K4: BN -> Wa projection -> BN -> +residual -> PReLU (per (n,t)) ======
__global__ void k4_out(const bf16_t* __restrict__ aobuf,
                       const float*  __restrict__ x,
                       const float* __restrict__ cg, const float* __restrict__ cb,
                       const float* __restrict__ cm, const float* __restrict__ cv,
                       const bf16_t* __restrict__ fWag, const float* __restrict__ ba,
                       const float* __restrict__ g2, const float* __restrict__ b2,
                       const float* __restrict__ m2, const float* __restrict__ v2,
                       const float* __restrict__ pout,
                       float* __restrict__ out) {
  __shared__ __align__(32) bf16_t fFS[4 * 512];
  __shared__ __align__(16) float s1S[Cc], d1S[Cc], s2S[Cc], d2S[Cc];

  int tid = threadIdx.x;
  int nt  = blockIdx.x;
  int n = nt / Tt, t = nt % Tt;

  if (tid < Cc) {
    float s = cg[tid] * rsqrtf(cv[tid] + 1e-5f);
    s1S[tid] = s;  d1S[tid] = cb[tid] - cm[tid] * s;
    float s2 = g2[tid] * rsqrtf(v2[tid] + 1e-5f);
    s2S[tid] = s2; d2S[tid] = s2 * (ba[tid] - m2[tid]) + b2[tid];
  }
  __syncthreads();

  // stage BN1(ao) -> B-frag layout, guard hoisted to octet granularity
  for (int i = tid; i < 8 * 32; i += 256) {
    int o = i >> 5, v = i & 31, c0 = o * 8;
    v8bf pk = {};
    if (v < Vv) {
      float sv[8], dv[8];
      ld8f(s1S + c0, sv);
      ld8f(d1S + c0, dv);
#pragma unroll
      for (int r = 0; r < 8; ++r)
        pk[r] = to_bf16(sv[r] * from_bf16(aobuf[((size_t)nt * Cc + c0 + r) * Vv + v]) + dv[r]);
    }
    *(v8bf*)(fFS + b_oct(c0, v, 2)) = pk;
  }
  __syncthreads();

  float ap = pout[0];
  int wave = tid >> 5, l = tid & 31;
  int mt = wave >> 1, ntile = wave & 1;
  int hb = 8 * (l >> 4);
  {
    v8f acc = {};
#pragma unroll
    for (int ks = 0; ks < 2; ++ks) {
      v16bf a = ldfrag(fWag + ((mt * 2 + ks) * 32 + l) * 16);
      v16bf b = ldfrag(fFS + ((ks * 2 + ntile) * 32 + l) * 16);
      acc = wmma_bf16(a, b, acc);
    }
    int col = ntile * 16 + (l & 15);
    if (col < Vv) {
      float sv[8], dv[8];
      ld8f(s2S + mt * 16 + hb, sv);
      ld8f(d2S + mt * 16 + hb, dv);
#pragma unroll
      for (int r = 0; r < 8; ++r) {
        int oc = mt * 16 + hb + r;
        float y = sv[r] * acc[r] + dv[r];
        float res = x[(((size_t)n * Cc + oc) * Tt + t) * Vv + col];
        float z = y + res;
        out[(((size_t)n * Cc + oc) * Tt + t) * Vv + col] = (z >= 0.f) ? z : ap * z;
      }
    }
  }
}

// --------------------------- launch ---------------------------
static inline size_t align256(size_t s) { return (s + 255) & ~(size_t)255; }

extern "C" void kernel_launch(void* const* d_in, const int* in_sizes, int n_in,
                              void* d_out, int out_size, void* d_ws, size_t ws_size,
                              hipStream_t stream) {
  (void)in_sizes; (void)n_in; (void)out_size; (void)ws_size;
  const float* x    = (const float*)d_in[0];
  const float* Aw   = (const float*)d_in[1];
  const float* Wt0  = (const float*)d_in[2];
  const float* bt0  = (const float*)d_in[3];
  const float* g0   = (const float*)d_in[4];
  const float* be0  = (const float*)d_in[5];
  const float* m0   = (const float*)d_in[6];
  const float* v0   = (const float*)d_in[7];
  const float* p0   = (const float*)d_in[8];
  const float* Wt1  = (const float*)d_in[9];
  const float* bt1  = (const float*)d_in[10];
  const float* g1   = (const float*)d_in[11];
  const float* be1  = (const float*)d_in[12];
  const float* m1   = (const float*)d_in[13];
  const float* v1   = (const float*)d_in[14];
  const float* p1   = (const float*)d_in[15];
  const float* Wqkv = (const float*)d_in[16];
  const float* bqkv = (const float*)d_in[17];
  const float* cg   = (const float*)d_in[18];
  const float* cb   = (const float*)d_in[19];
  const float* cm   = (const float*)d_in[20];
  const float* cv   = (const float*)d_in[21];
  const float* Wa   = (const float*)d_in[22];
  const float* ba   = (const float*)d_in[23];
  const float* g2   = (const float*)d_in[24];
  const float* b2   = (const float*)d_in[25];
  const float* m2   = (const float*)d_in[26];
  const float* v2   = (const float*)d_in[27];
  const float* pout = (const float*)d_in[28];

  char* ws = (char*)d_ws;
  size_t off = 0;
  bf16_t* fW0 = (bf16_t*)(ws + off); off += align256((size_t)HID * Cc  * 2);
  bf16_t* fW1 = (bf16_t*)(ws + off); off += align256((size_t)HID * HID * 2);
  bf16_t* fWq = (bf16_t*)(ws + off); off += align256((size_t)QC  * Cc  * 2);
  bf16_t* fWa = (bf16_t*)(ws + off); off += align256((size_t)Cc  * Cc  * 2);
  bf16_t* hbuf   = (bf16_t*)(ws + off); off += align256((size_t)Nn * Tt * Vv * HID * 2);
  bf16_t* qkvbuf = (bf16_t*)(ws + off); off += align256((size_t)Nn * Vv * QC * Tt * 2);
  bf16_t* aobuf  = (bf16_t*)(ws + off);

  k0_prep<<<48, 256, 0, stream>>>(Wt0, Wt1, Wqkv, Wa, fW0, fW1, fWq, fWa);
  k1_mlp<<<Nn * Tt, 128, 0, stream>>>(x, fW0, fW1, bt0, g0, be0, m0, v0, p0,
                                      bt1, g1, be1, m1, v1, p1, hbuf);
  k2_qkv<<<Nn * Tt, 256, 0, stream>>>(hbuf, Aw, fWq, bqkv, qkvbuf);
  k3_attn<<<Nn * Vv * HH, 32, 0, stream>>>(qkvbuf, aobuf);
  k4_out<<<Nn * Tt, 256, 0, stream>>>(aobuf, x, cg, cb, cm, cv, fWa, ba,
                                      g2, b2, m2, v2, pout, (float*)d_out);
}
